// RadianceField_11227044512351
// MI455X (gfx1250) — matrix operand order repaired and checked by reading, not compiled, not run on hardware
//
#include <hip/hip_runtime.h>
#include <stdint.h>

#define IDIM 192
#define NVOX (IDIM * IDIM * IDIM)
#define NB_SAMPLES 128
#define NB_RAYS 16384
#define RAYS_PER_BLOCK 64
#define THREADS 128
#define CSTRIDE 132  // padded row stride to spread LDS banks

typedef __bf16 bf16x16 __attribute__((ext_vector_type(16)));
typedef float  f32x8   __attribute__((ext_vector_type(8)));

// ---------------------------------------------------------------------------
// Deterministic per-(ray,sample) uniform in [0,1)
// ---------------------------------------------------------------------------
__device__ __forceinline__ float rnd01(uint32_t x) {
  x ^= x >> 17; x *= 0xed5ad4bbu;
  x ^= x >> 11; x *= 0xac4c1b51u;
  x ^= x >> 15; x *= 0x31848babu;
  x ^= x >> 14;
  return (float)(x >> 8) * (1.0f / 16777216.0f);
}

// ---------------------------------------------------------------------------
// Kernel 1: fuse grid.  f.x = sum over 9 SH channels, f.y = opacity.
// One-time 283MB stream -> 57MB table that is L2-resident afterwards.
// (sigmoid(sum_c harm_c): channel-sum commutes with trilinear interp)
// ---------------------------------------------------------------------------
__global__ void fuse_grid_kernel(const float* __restrict__ grid,
                                 const float* __restrict__ opacity,
                                 float2* __restrict__ fused) {
  int v = blockIdx.x * blockDim.x + threadIdx.x;
  if (v >= NVOX) return;
  const float* g = grid + (size_t)v * 9;
  float s = 0.f;
#pragma unroll
  for (int c = 0; c < 9; ++c) s += g[c];
  fused[v] = make_float2(s, opacity[v]);
}

// ---------------------------------------------------------------------------
// Kernel 2: ray-march.  64 rays / block, 128 threads (4 waves).
// Phase 1: per-sample trilinear gathers from the fused table (8 x b64, L2).
// Phase 2: per-wave exclusive prefix-scan of cur as a triangular matmul
//          E(16x128) = CUR(16x128) x U(128x128), U[k,n]=(k<n), done with
//          20 v_wmma_f32_16x16x32_bf16 per wave (zero blocks skipped,
//          diagonal masks hoisted), then composite with register
//          accumulation + shfl_xor tree reduction.
// ---------------------------------------------------------------------------
__global__ __launch_bounds__(THREADS)
void raymarch_kernel(const float* __restrict__ xin,
                     const float* __restrict__ din,
                     const float2* __restrict__ fused,
                     float* __restrict__ out) {
  __shared__ __bf16 curS[RAYS_PER_BLOCK][CSTRIDE];   // per-step opacity (bf16)
  __shared__ float  mvalS[RAYS_PER_BLOCK][CSTRIDE];  // (1-exp(-cur))*sigmoid
  __shared__ float  rox[RAYS_PER_BLOCK][3];
  __shared__ float  rdi[RAYS_PER_BLOCK][3];
  __shared__ float  rtm[RAYS_PER_BLOCK][2];

  const int tid   = threadIdx.x;
  const int rbase = blockIdx.x * RAYS_PER_BLOCK;

  // ---- stage ray data + ray/AABB intersection -----------------------------
  if (tid < RAYS_PER_BLOCK) {
    const int g = rbase + tid;
    const float INF0 = (float)IDIM * (float)IDIM * (float)IDIM;
    float tmn = -INF0, tmx = INF0;
#pragma unroll
    for (int a = 0; a < 3; ++a) {
      float xa = xin[g * 3 + a];
      float da = din[g * 3 + a];
      rox[tid][a] = xa;
      rdi[tid][a] = da;
      float invd = 1.0f / da;
      float t0 = (0.0f - xa) * invd;
      float t1 = ((float)(IDIM - 1) - xa) * invd;
      tmn = fmaxf(tmn, fminf(t0, t1));
      tmx = fminf(tmx, fmaxf(t0, t1));
    }
    rtm[tid][0] = tmn;
    rtm[tid][1] = tmx;
  }
  __syncthreads();

  // ---- phase 1: all 128 threads march one ray per iteration ---------------
  // thread tid <-> sample tid  (consecutive lanes = consecutive samples:
  // spatially adjacent points -> L2-resident, coherent gathers)
  for (int r = 0; r < RAYS_PER_BLOCK; ++r) {
    const int s     = tid;
    const float tmn = rtm[r][0];
    const float dt  = (rtm[r][1] - tmn) * (1.0f / NB_SAMPLES);
    const uint32_t gid = (uint32_t)(rbase + r) * (uint32_t)NB_SAMPLES;

    // stratified jittered samples: monotone by construction (no sort needed)
    float u0 = rnd01(gid + (uint32_t)s);
    float u1 = rnd01(gid + (uint32_t)s + 1u);
    float t     = tmn + ((float)s + u0) * dt;
    float tnext = tmn + ((float)(s + 1) + u1) * dt;
    float delta = tnext - t;

    float px = rox[r][0] + t * rdi[r][0];
    float py = rox[r][1] + t * rdi[r][1];
    float pz = rox[r][2] + t * rdi[r][2];

    int ix = min(max((int)floorf(px), 0), IDIM - 2);
    int iy = min(max((int)floorf(py), 0), IDIM - 2);
    int iz = min(max((int)floorf(pz), 0), IDIM - 2);
    float wx = px - (float)ix;
    float wy = py - (float)iy;
    float wz = pz - (float)iz;

    const int base = (ix * IDIM + iy) * IDIM + iz;
    float ssum = 0.0f, op = 0.0f;
#pragma unroll
    for (int ox = 0; ox < 2; ++ox) {
      float wox = ox ? wx : (1.0f - wx);
#pragma unroll
      for (int oy = 0; oy < 2; ++oy) {
        float woy = oy ? wy : (1.0f - wy);
#pragma unroll
        for (int oz = 0; oz < 2; ++oz) {
          float w = wox * woy * (oz ? wz : (1.0f - wz));
          float2 f = fused[base + ox * (IDIM * IDIM) + oy * IDIM + oz];
          ssum += w * f.x;
          op   += w * f.y;
        }
      }
    }

    const bool live = (s < NB_SAMPLES - 1);
    float cur = live ? delta * op : 0.0f;
    float col = 1.0f / (1.0f + __expf(-ssum));            // sigmoid
    float mv  = live ? (1.0f - __expf(-cur)) * col : 0.0f;
    curS[r][s]  = (__bf16)cur;
    mvalS[r][s] = mv;
  }
  __syncthreads();

  // ---- phase 2: WMMA exclusive scan + compositing -------------------------
  const int lane = tid & 31;
  const int wave = tid >> 5;       // 0..3
  const int l16  = lane & 15;
  const int hi   = lane >> 4;      // 0/1
  const int R0   = wave * 16;      // this wave's first ray (local)

  // Hoisted B fragments (B order used here, matching A: K = e + 16*hi, N = l16):
  //   B1      : all-ones block (K-block entirely below diagonal)
  //   Bd_even : diagonal block mask for even j  -> (hi==0 && e < l16)
  //   Bd_odd  : diagonal block mask for odd  j  -> (hi==0 || e < l16)
  bf16x16 B1, Bd_even, Bd_odd;
#pragma unroll
  for (int e = 0; e < 16; ++e) {
    B1[e]      = (__bf16)1.0f;
    Bd_even[e] = (hi == 0 && e < l16) ? (__bf16)1.0f : (__bf16)0.0f;
    Bd_odd[e]  = (hi == 0 || e < l16) ? (__bf16)1.0f : (__bf16)0.0f;
  }

  // Hoisted A fragments (16-bit A layout), one per 32-sample K-block:
  bf16x16 A[4];
#pragma unroll
  for (int kb = 0; kb < 4; ++kb) {
#pragma unroll
    for (int e = 0; e < 16; ++e) {
      int K = 16 * hi + e;
      A[kb][e] = curS[R0 + l16][32 * kb + K];
    }
  }

  float partial[8];
#pragma unroll
  for (int v = 0; v < 8; ++v) partial[v] = 0.0f;

#pragma unroll
  for (int j = 0; j < 8; ++j) {    // output column blocks of 16 samples
    f32x8 D = {0.f, 0.f, 0.f, 0.f, 0.f, 0.f, 0.f, 0.f};
    const int kbd = j >> 1;        // diagonal K-block index
#pragma unroll
    for (int kb = 0; kb < kbd; ++kb) {   // fully-below-diagonal blocks
      D = __builtin_amdgcn_wmma_f32_16x16x32_bf16(
          false, A[kb], false, B1, (short)0, D, false, false);
    }
    D = __builtin_amdgcn_wmma_f32_16x16x32_bf16(
        false, A[kbd], false, (j & 1) ? Bd_odd : Bd_even,
        (short)0, D, false, false);

    // D[v] = E[m = v + 8*hi][n = 16*j + l16]  (exclusive cumsum of cur)
#pragma unroll
    for (int v = 0; v < 8; ++v) {
      partial[v] += __expf(-D[v]) * mvalS[R0 + v + 8 * hi][16 * j + l16];
    }
  }

  // cross-lane reduction over the 16 N-lanes of each half; rows are unique
  // per (wave, v, hi), so plain global stores (no atomics, no LDS round-trip)
#pragma unroll
  for (int v = 0; v < 8; ++v) {
    float p = partial[v];
    p += __shfl_xor(p, 1, 16);
    p += __shfl_xor(p, 2, 16);
    p += __shfl_xor(p, 4, 16);
    p += __shfl_xor(p, 8, 16);
    if (l16 == 0) out[rbase + R0 + v + 8 * hi] = p;
  }
}

// ---------------------------------------------------------------------------
// Host entry
// ---------------------------------------------------------------------------
extern "C" void kernel_launch(void* const* d_in, const int* in_sizes, int n_in,
                              void* d_out, int out_size, void* d_ws, size_t ws_size,
                              hipStream_t stream) {
  const float* x       = (const float*)d_in[0];   // [16384,3]
  const float* d       = (const float*)d_in[1];   // [16384,3]
  const float* grid    = (const float*)d_in[2];   // [192^3, 9]
  const float* opacity = (const float*)d_in[3];   // [192^3]
  float*       out     = (float*)d_out;           // [16384]
  float2*      fused   = (float2*)d_ws;           // [192^3] {sum9, opacity}

  (void)in_sizes; (void)n_in; (void)out_size; (void)ws_size;

  fuse_grid_kernel<<<(NVOX + 255) / 256, 256, 0, stream>>>(grid, opacity, fused);
  raymarch_kernel<<<NB_RAYS / RAYS_PER_BLOCK, THREADS, 0, stream>>>(x, d, fused, out);
}
